// SweepStereo_32950989095290
// MI455X (gfx1250) — matrix operand order repaired and checked by reading, not compiled, not run on hardware
//
#include <hip/hip_runtime.h>
#include <hip/hip_bf16.h>

typedef _Float16 half_t;
typedef __attribute__((ext_vector_type(16))) _Float16 v16h;
typedef __attribute__((ext_vector_type(8)))  _Float16 v8h;
typedef __attribute__((ext_vector_type(8)))  float    v8f;

// D = A(16 out-ch x 32 K) x B(32 K x 16 pixels) + C, f32 accum
__device__ __forceinline__ v8f wmma_f16(v16h a, v16h b, v8f c) {
  return __builtin_amdgcn_wmma_f32_16x16x32_f16(false, a, false, b, (short)0, c, false, false);
}

__device__ __forceinline__ float sigmoid_f(float x) { return 1.0f / (1.0f + expf(-x)); }

__device__ __forceinline__ v16h zero_v16h() { v16h r = {}; return r; }

// Halo-padded GRU window buffers: rows y in [-1,96], cols x in [-1,160], 32ch fp16
#define PAD_HALVES ((size_t)98 * 162 * 32)
__device__ __forceinline__ size_t pidx(int y, int x) {
  return ((size_t)(y + 1) * 162 + (x + 1)) * 32;
}
// Halo-padded encoder stage-1 activations: rows [-1,192], cols [-1,320], 32ch fp16
#define PAD1_HALVES ((size_t)194 * 322 * 32)
__device__ __forceinline__ size_t pidx1(int y, int x) {
  return ((size_t)(y + 1) * 322 + (x + 1)) * 32;
}

// ---------------------------------------------------------------------------
// Weight repack: fp32 OIHW -> fp16 A-fragment lane order (wave32).
// lane = (k_hi<<4)|m ; slot j: K = j + 16*k_hi
// K-chunk c = tap * (Cin/32) + cin_block.
// Packed elem: ((mt*KchTot + cOff + c)*32 + lane)*16 + j
// ---------------------------------------------------------------------------
__global__ void pack_w_kernel(const float* __restrict__ W, half_t* __restrict__ P,
                              int M, int Cin, int KH, int KW, int KchTot, int cOff) {
  int G = Cin >> 5;
  int Kloc = KH * KW * G;
  int Mt = M >> 4;
  int total = Mt * Kloc * 512;
  int idx = blockIdx.x * 256 + threadIdx.x;
  if (idx >= total) return;
  int j    = idx & 15;
  int lane = (idx >> 4) & 31;
  int c    = (idx >> 9) % Kloc;
  int mt   = (idx >> 9) / Kloc;
  int m  = mt * 16 + (lane & 15);
  int hi = lane >> 4;
  int ks = j + 16 * hi;
  int t = c / G, g = c % G;
  int cin = g * 32 + ks;
  int kh = t / KW, kw = t % KW;
  float v = W[((m * Cin + cin) * KH + kh) * KW + kw];
  P[(((size_t)mt * KchTot + cOff + c) * 32 + lane) * 16 + j] = (half_t)v;
}

__global__ void zero_kernel(uint4* __restrict__ p, int n4) {
  int i = blockIdx.x * 256 + threadIdx.x;
  if (i < n4) { uint4 z; z.x = z.y = z.z = z.w = 0u; p[i] = z; }
}

// ---------------------------------------------------------------------------
// Entry conv: 7x7 s2 p3, 3->32 ch, fused 2*(x/255)-1. fp32 planes out.
// ---------------------------------------------------------------------------
__global__ void conv7_s2_kernel(const float* __restrict__ img, const float* __restrict__ W,
                                const float* __restrict__ b, float* __restrict__ out,
                                int Hi, int Wi, int Ho, int Wo) {
  int idx = blockIdx.x * 256 + threadIdx.x;
  int total = 32 * Ho * Wo;
  if (idx >= total) return;
  int x = idx % Wo;
  int y = (idx / Wo) % Ho;
  int co = idx / (Wo * Ho);
  float acc = b[co];
  const float k = 2.0f / 255.0f;
  for (int ci = 0; ci < 3; ++ci)
    for (int ky = 0; ky < 7; ++ky) {
      int iy = 2 * y + ky - 3;
      if (iy < 0 || iy >= Hi) continue;
      for (int kx = 0; kx < 7; ++kx) {
        int ix = 2 * x + kx - 3;
        if (ix < 0 || ix >= Wi) continue;
        float v = img[(ci * Hi + iy) * Wi + ix] * k - 1.0f;
        acc += v * W[((co * 3 + ci) * 7 + ky) * 7 + kx];
      }
    }
  out[(co * Ho + y) * Wo + x] = acc;
}

// ---------------------------------------------------------------------------
// Instance norm (per channel) + ReLU: fp32 planes -> fp16 [...][32] with
// optional halo padding: out pixel (y,x) -> ((y+pad)*ostride + (x+pad))*32 + c
// ---------------------------------------------------------------------------
__global__ void inorm_relu_kernel(const float* __restrict__ in, half_t* __restrict__ out,
                                  int HW, int Wrow, int ostride, int pad) {
  __shared__ float ss[256], sq[256];
  int c = blockIdx.x;
  int tid = threadIdx.x;
  const float* p = in + (size_t)c * HW;
  float s = 0.f, q = 0.f;
  for (int i = tid; i < HW; i += 256) { float v = p[i]; s += v; q += v * v; }
  ss[tid] = s; sq[tid] = q;
  __syncthreads();
  for (int o = 128; o > 0; o >>= 1) {
    if (tid < o) { ss[tid] += ss[tid + o]; sq[tid] += sq[tid + o]; }
    __syncthreads();
  }
  float mean = ss[0] / (float)HW;
  float var  = sq[0] / (float)HW - mean * mean;
  float rs = rsqrtf(var + 1e-5f);
  for (int i = tid; i < HW; i += 256) {
    float v = (p[i] - mean) * rs;
    int y = i / Wrow, x = i % Wrow;
    out[((size_t)(y + pad) * ostride + (x + pad)) * 32 + c] = (half_t)fmaxf(v, 0.0f);
  }
}

// ---------------------------------------------------------------------------
// 3x3 s2 p1 conv, 32->32, WMMA; halo-padded fp16 input (194x322), A from LDS;
// fp32 planes out (+bias) for the next instance norm.
// ---------------------------------------------------------------------------
__global__ void conv2_s2_wmma_kernel(const half_t* __restrict__ actp,
                                     const half_t* __restrict__ P,
                                     const float* __restrict__ bias,
                                     float* __restrict__ out,
                                     int Ho, int Wo) {
  extern __shared__ half_t sA[];
  int tid = threadIdx.x;
  {
    const uint4* srcp = (const uint4*)P;
    uint4* dstp = (uint4*)sA;
    for (int i = tid; i < 2 * 9 * 64; i += 256) dstp[i] = srcp[i];
  }
  __syncthreads();
  int wid = blockIdx.x * 8 + (tid >> 5);
  int XT = Wo >> 4;
  bool actv = wid < Ho * XT;
  int w2 = actv ? wid : 0;
  int y = w2 / XT, x0 = (w2 % XT) << 4;
  int lane = tid & 31, n = lane & 15, hi = lane >> 4;
  v8f acc0 = {}; v8f acc1 = {};
  for (int c = 0; c < 9; ++c) {
    int dy = c / 3 - 1, dx = c % 3 - 1;
    int iy = 2 * y + dy, ix = 2 * (x0 + n) + dx;
    v16h b = *(const v16h*)(actp + pidx1(iy, ix) + hi * 16);
    v16h a0 = *(const v16h*)(sA + ((size_t)(0 * 9 + c) * 32 + lane) * 16);
    v16h a1 = *(const v16h*)(sA + ((size_t)(1 * 9 + c) * 32 + lane) * 16);
    acc0 = wmma_f16(a0, b, acc0);
    acc1 = wmma_f16(a1, b, acc1);
  }
  if (actv) {
    int x = x0 + n;
    #pragma unroll
    for (int j = 0; j < 8; ++j) {
      int c0 = j + 8 * hi;
      out[((size_t)c0 * Ho + y) * Wo + x]        = acc0[j] + bias[c0];
      out[((size_t)(c0 + 16) * Ho + y) * Wo + x] = acc1[j] + bias[c0 + 16];
    }
  }
}

// ---------------------------------------------------------------------------
// 1x1 conv 32->32 + bias. mode 0: fp16 out (fmap). mode 1: fp32 tanh (hidden).
// ---------------------------------------------------------------------------
__global__ void conv1x1_wmma_kernel(const half_t* __restrict__ act,
                                    const half_t* __restrict__ P,
                                    const float* __restrict__ bias,
                                    half_t* __restrict__ out16,
                                    float* __restrict__ out32,
                                    int HW, int mode) {
  int wid = blockIdx.x * 8 + (threadIdx.x >> 5);
  if (wid >= (HW >> 4)) return;
  int p0 = wid << 4;
  int lane = threadIdx.x & 31, n = lane & 15, hi = lane >> 4;
  v16h b = *(const v16h*)(act + ((size_t)(p0 + n)) * 32 + hi * 16);
  v8f acc0 = {}; v8f acc1 = {};
  v16h a0 = *(const v16h*)(P + ((size_t)0 * 32 + lane) * 16);
  v16h a1 = *(const v16h*)(P + ((size_t)32 + lane) * 16);
  acc0 = wmma_f16(a0, b, acc0);
  acc1 = wmma_f16(a1, b, acc1);
  size_t base = (size_t)(p0 + n) * 32;
  if (mode == 0) {
    v8h e0, e1;
    #pragma unroll
    for (int j = 0; j < 8; ++j) {
      e0[j] = (half_t)(acc0[j] + bias[8 * hi + j]);
      e1[j] = (half_t)(acc1[j] + bias[16 + 8 * hi + j]);
    }
    *(v8h*)(out16 + base + 8 * hi)      = e0;
    *(v8h*)(out16 + base + 16 + 8 * hi) = e1;
  } else {
    v8f e0, e1;
    #pragma unroll
    for (int j = 0; j < 8; ++j) {
      e0[j] = tanhf(acc0[j] + bias[8 * hi + j]);
      e1[j] = tanhf(acc1[j] + bias[16 + 8 * hi + j]);
    }
    *(v8f*)(out32 + base + 8 * hi)      = e0;
    *(v8f*)(out32 + base + 16 + 8 * hi) = e1;
  }
}

// ---------------------------------------------------------------------------
// feat (per sweep step d): feat = A*f1 + B*f2 + CD*(f1.*f2) + b  (K=96 GEMM)
// Also copies hidden window to fp16 h16 (halo-padded). Writes zeros at the
// window's right edge so the 3x3 convs can load unconditionally.
// ---------------------------------------------------------------------------
__global__ void feat_wmma_kernel(const half_t* __restrict__ f1map,
                                 const half_t* __restrict__ f2map,
                                 const float* __restrict__ hidden,
                                 const half_t* __restrict__ Pfeat,
                                 const float* __restrict__ cd_b,
                                 half_t* __restrict__ h16,
                                 half_t* __restrict__ feat16, int d) {
  int Wd = 160 - d;
  int XT = (Wd >> 4) + 1;  // overshoot one tile to zero the edge column
  int wid = blockIdx.x * 8 + (threadIdx.x >> 5);
  if (wid >= 96 * XT) return;
  int y = wid / XT, x0 = (wid % XT) << 4;
  int lane = threadIdx.x & 31, n = lane & 15, hi = lane >> 4;
  int xp = x0 + n;
  bool valid = xp < Wd;
  int xc = valid ? xp : (Wd - 1);
  size_t src = ((size_t)y * 160 + xc) * 32 + hi * 16;
  v16h f1 = *(const v16h*)(f1map + src + (size_t)d * 32);
  v16h f2 = *(const v16h*)(f2map + src);
  v16h pr = f1 * f2;

  // hidden window -> fp16
  const float* hp = hidden + src + (size_t)d * 32;
  v8f ha = *(const v8f*)(hp);
  v8f hb = *(const v8f*)(hp + 8);
  v16h hv;
  #pragma unroll
  for (int i = 0; i < 8; ++i) { hv[i] = (half_t)ha[i]; hv[8 + i] = (half_t)hb[i]; }
  if (!valid) hv = zero_v16h();

  v8f acc0 = {}; v8f acc1 = {};
  {
    v16h a0 = *(const v16h*)(Pfeat + (0 * 32 + lane) * 16);
    v16h a1 = *(const v16h*)(Pfeat + ((3 + 0) * 32 + lane) * 16);
    acc0 = wmma_f16(a0, f1, acc0); acc1 = wmma_f16(a1, f1, acc1);
    a0 = *(const v16h*)(Pfeat + (1 * 32 + lane) * 16);
    a1 = *(const v16h*)(Pfeat + ((3 + 1) * 32 + lane) * 16);
    acc0 = wmma_f16(a0, f2, acc0); acc1 = wmma_f16(a1, f2, acc1);
    a0 = *(const v16h*)(Pfeat + (2 * 32 + lane) * 16);
    a1 = *(const v16h*)(Pfeat + ((3 + 2) * 32 + lane) * 16);
    acc0 = wmma_f16(a0, pr, acc0); acc1 = wmma_f16(a1, pr, acc1);
  }
  if (xp <= 160) {
    size_t pb = pidx(y, xp);
    *(v16h*)(h16 + pb + hi * 16) = hv;
    v8h e0, e1;
    #pragma unroll
    for (int j = 0; j < 8; ++j) {
      e0[j] = valid ? (half_t)(acc0[j] + cd_b[8 * hi + j]) : (half_t)0;
      e1[j] = valid ? (half_t)(acc1[j] + cd_b[16 + 8 * hi + j]) : (half_t)0;
    }
    *(v8h*)(feat16 + pb + 8 * hi)      = e0;
    *(v8h*)(feat16 + pb + 16 + 8 * hi) = e1;
  }
}

// ---------------------------------------------------------------------------
// 3x3 Cin=96 conv accumulation, A fragments from LDS, halo-padded inputs so
// all B loads are unconditional.
// ---------------------------------------------------------------------------
template <int MT>
__device__ __forceinline__ void conv3x3_lds(const half_t* __restrict__ b0,
                                            const half_t* __restrict__ b1,
                                            const half_t* __restrict__ b2,
                                            const half_t* __restrict__ sA,
                                            int y, int xc, int hi, int lane,
                                            v8f acc[MT]) {
  for (int t = 0; t < 9; ++t) {
    int dy = t / 3 - 1, dx = t % 3 - 1;
    size_t off = pidx(y + dy, xc + dx) + hi * 16;
    #pragma unroll
    for (int blk = 0; blk < 3; ++blk) {
      const half_t* bp = (blk == 0) ? b0 : ((blk == 1) ? b1 : b2);
      v16h b = *(const v16h*)(bp + off);
      int c = t * 3 + blk;
      #pragma unroll
      for (int mt = 0; mt < MT; ++mt) {
        v16h a = *(const v16h*)(sA + (((size_t)mt * 27 + c) * 32 + lane) * 16);
        acc[mt] = wmma_f16(a, b, acc[mt]);
      }
    }
  }
}

// z||r conv over hx=[h,h,feat], Cout=64; sigmoid; rh = r.*h
__global__ void zr_wmma_kernel(const half_t* __restrict__ h16,
                               const half_t* __restrict__ feat16,
                               const half_t* __restrict__ Pzr,
                               const float* __restrict__ bz,
                               const float* __restrict__ br,
                               float* __restrict__ z32,
                               half_t* __restrict__ rh16, int d) {
  extern __shared__ half_t sA[];
  int tid = threadIdx.x;
  {
    const uint4* srcp = (const uint4*)Pzr;
    uint4* dstp = (uint4*)sA;
    for (int i = tid; i < 4 * 27 * 64; i += 256) dstp[i] = srcp[i];
  }
  __syncthreads();
  int Wd = 160 - d;
  int XT = (Wd >> 4) + 1;
  int wid = blockIdx.x * 8 + (tid >> 5);
  bool act = wid < 96 * XT;
  int w2 = act ? wid : 0;
  int y = w2 / XT, x0 = (w2 % XT) << 4;
  int lane = tid & 31, n = lane & 15, hi = lane >> 4;
  int xp = x0 + n;
  int xc = xp < Wd ? xp : (Wd - 1);
  v8f acc[4]; acc[0] = {}; acc[1] = {}; acc[2] = {}; acc[3] = {};
  conv3x3_lds<4>(h16, h16, feat16, sA, y, xc, hi, lane, acc);
  if (act && xp <= 160) {
    bool valid = xp < Wd;
    size_t pb = pidx(y, xp);
    size_t pc = pidx(y, xc);
    v8h hA = *(const v8h*)(h16 + pc + 8 * hi);
    v8h hB = *(const v8h*)(h16 + pc + 16 + 8 * hi);
    v8f zv0, zv1;
    v8h r0v, r1v;
    #pragma unroll
    for (int j = 0; j < 8; ++j) {
      zv0[j] = sigmoid_f(acc[0][j] + bz[8 * hi + j]);
      zv1[j] = sigmoid_f(acc[1][j] + bz[16 + 8 * hi + j]);
      float r0 = sigmoid_f(acc[2][j] + br[8 * hi + j]);
      float r1 = sigmoid_f(acc[3][j] + br[16 + 8 * hi + j]);
      r0v[j] = valid ? (half_t)(r0 * (float)hA[j]) : (half_t)0;
      r1v[j] = valid ? (half_t)(r1 * (float)hB[j]) : (half_t)0;
    }
    if (valid) {
      size_t zb = ((size_t)y * 160 + xp) * 32;
      *(v8f*)(z32 + zb + 8 * hi)      = zv0;
      *(v8f*)(z32 + zb + 16 + 8 * hi) = zv1;
    }
    *(v8h*)(rh16 + pb + 8 * hi)      = r0v;
    *(v8h*)(rh16 + pb + 16 + 8 * hi) = r1v;
  }
}

// q conv over [r*h, h, feat]; GRU update into fp32 hidden
__global__ void q_wmma_kernel(const half_t* __restrict__ rh16,
                              const half_t* __restrict__ h16,
                              const half_t* __restrict__ feat16,
                              const half_t* __restrict__ Pq,
                              const float* __restrict__ bq,
                              const float* __restrict__ z32,
                              float* __restrict__ hidden, int d) {
  extern __shared__ half_t sA[];
  int tid = threadIdx.x;
  {
    const uint4* srcp = (const uint4*)Pq;
    uint4* dstp = (uint4*)sA;
    for (int i = tid; i < 2 * 27 * 64; i += 256) dstp[i] = srcp[i];
  }
  __syncthreads();
  int Wd = 160 - d;
  int XT = (Wd + 15) >> 4;
  int wid = blockIdx.x * 8 + (tid >> 5);
  bool act = wid < 96 * XT;
  int w2 = act ? wid : 0;
  int y = w2 / XT, x0 = (w2 % XT) << 4;
  int lane = tid & 31, n = lane & 15, hi = lane >> 4;
  int xp = x0 + n;
  int xc = xp < Wd ? xp : (Wd - 1);
  v8f acc[2]; acc[0] = {}; acc[1] = {};
  conv3x3_lds<2>(rh16, h16, feat16, sA, y, xc, hi, lane, acc);
  if (act && xp < Wd) {
    size_t zb = ((size_t)y * 160 + xp) * 32;
    size_t hb = ((size_t)y * 160 + xp + d) * 32;
    v8f z0 = *(const v8f*)(z32 + zb + 8 * hi);
    v8f z1 = *(const v8f*)(z32 + zb + 16 + 8 * hi);
    v8f h0 = *(const v8f*)(hidden + hb + 8 * hi);
    v8f h1 = *(const v8f*)(hidden + hb + 16 + 8 * hi);
    v8f o0, o1;
    #pragma unroll
    for (int j = 0; j < 8; ++j) {
      float q0 = tanhf(acc[0][j] + bq[8 * hi + j]);
      float q1 = tanhf(acc[1][j] + bq[16 + 8 * hi + j]);
      o0[j] = (1.0f - z0[j]) * h0[j] + z0[j] * q0;
      o1[j] = (1.0f - z1[j]) * h1[j] + z1[j] * q1;
    }
    *(v8f*)(hidden + hb + 8 * hi)      = o0;
    *(v8f*)(hidden + hb + 16 + 8 * hi) = o1;
  }
}

// hidden fp32 -> halo-padded fp16
__global__ void cvt_h16_kernel(const float* __restrict__ src, half_t* __restrict__ dst) {
  int i = blockIdx.x * 256 + threadIdx.x;
  if (i >= 96 * 160 * 32) return;
  int c = i & 31;
  int x = (i >> 5) % 160;
  int y = (i >> 5) / 160;
  dst[pidx(y, x) + c] = (half_t)src[i];
}

// disparity head conv1: 3x3 p1, 32->64, relu, fp16 out [HW][64]; A from LDS
__global__ void disp1_wmma_kernel(const half_t* __restrict__ actp,  // halo-padded
                                  const half_t* __restrict__ P,
                                  const float* __restrict__ bias,
                                  half_t* __restrict__ out) {
  extern __shared__ half_t sA[];
  int tid = threadIdx.x;
  {
    const uint4* srcp = (const uint4*)P;
    uint4* dstp = (uint4*)sA;
    for (int i = tid; i < 4 * 9 * 64; i += 256) dstp[i] = srcp[i];
  }
  __syncthreads();
  int wid = blockIdx.x * 8 + (tid >> 5);
  bool actv = wid < 96 * 10;
  int w2 = actv ? wid : 0;
  int y = w2 / 10, x0 = (w2 % 10) << 4;
  int lane = tid & 31, n = lane & 15, hi = lane >> 4;
  int x = x0 + n;
  v8f acc[4]; acc[0] = {}; acc[1] = {}; acc[2] = {}; acc[3] = {};
  for (int t = 0; t < 9; ++t) {
    int dy = t / 3 - 1, dx = t % 3 - 1;
    v16h b = *(const v16h*)(actp + pidx(y + dy, x + dx) + hi * 16);
    #pragma unroll
    for (int mt = 0; mt < 4; ++mt) {
      v16h a = *(const v16h*)(sA + (((size_t)mt * 9 + t) * 32 + lane) * 16);
      acc[mt] = wmma_f16(a, b, acc[mt]);
    }
  }
  if (actv) {
    size_t base = ((size_t)y * 160 + x) * 64;
    #pragma unroll
    for (int mt = 0; mt < 4; ++mt) {
      v8h e;
      #pragma unroll
      for (int j = 0; j < 8; ++j)
        e[j] = (half_t)fmaxf(acc[mt][j] + bias[mt * 16 + 8 * hi + j], 0.0f);
      *(v8h*)(out + base + mt * 16 + 8 * hi) = e;
    }
  }
}

// disparity head conv2: 3x3 p1, 64->1, negate -> fp32 d_out
__global__ void disp2_kernel(const half_t* __restrict__ act,
                             const float* __restrict__ W,
                             const float* __restrict__ b,
                             float* __restrict__ out) {
  int idx = blockIdx.x * 256 + threadIdx.x;
  if (idx >= 96 * 160) return;
  int x = idx % 160, y = idx / 160;
  float acc = b[0];
  for (int ky = 0; ky < 3; ++ky) {
    int iy = y + ky - 1;
    if (iy < 0 || iy >= 96) continue;
    for (int kx = 0; kx < 3; ++kx) {
      int ix = x + kx - 1;
      if (ix < 0 || ix >= 160) continue;
      const half_t* p = act + ((size_t)iy * 160 + ix) * 64;
      for (int c = 0; c < 64; ++c)
        acc += (float)p[c] * W[(c * 3 + ky) * 3 + kx];
    }
  }
  out[idx] = -acc;
}

// ---------------------------------------------------------------------------
extern "C" void kernel_launch(void* const* d_in, const int* in_sizes, int n_in,
                              void* d_out, int out_size, void* d_ws, size_t ws_size,
                              hipStream_t stream) {
  (void)in_sizes; (void)n_in; (void)out_size; (void)ws_size;
  const float* image1   = (const float*)d_in[0];
  const float* image2   = (const float*)d_in[1];
  const float* fnet_w1  = (const float*)d_in[2];
  const float* fnet_b1  = (const float*)d_in[3];
  const float* fnet_w2  = (const float*)d_in[4];
  const float* fnet_b2  = (const float*)d_in[5];
  const float* fnet_w3  = (const float*)d_in[6];
  const float* fnet_b3  = (const float*)d_in[7];
  const float* cnet_w1  = (const float*)d_in[8];
  const float* cnet_b1  = (const float*)d_in[9];
  const float* cnet_w2  = (const float*)d_in[10];
  const float* cnet_b2  = (const float*)d_in[11];
  const float* cnet_w3  = (const float*)d_in[12];
  const float* cnet_b3  = (const float*)d_in[13];
  const float* bform_a  = (const float*)d_in[14];
  const float* bform_b  = (const float*)d_in[15];
  const float* bform_cd = (const float*)d_in[16];
  const float* cd_b     = (const float*)d_in[17];
  const float* gru_wz   = (const float*)d_in[18];
  const float* gru_bz   = (const float*)d_in[19];
  const float* gru_wr   = (const float*)d_in[20];
  const float* gru_br   = (const float*)d_in[21];
  const float* gru_wq   = (const float*)d_in[22];
  const float* gru_bq   = (const float*)d_in[23];
  const float* disp_w1  = (const float*)d_in[24];
  const float* disp_b1  = (const float*)d_in[25];
  const float* disp_w2  = (const float*)d_in[26];
  const float* disp_b2  = (const float*)d_in[27];

  uint8_t* ws = (uint8_t*)d_ws;
  size_t off = 0;
  auto alloc = [&](size_t bytes) -> void* {
    void* p = ws + off;
    off = (off + bytes + 255) & ~(size_t)255;
    return p;
  };

  half_t* P_f2   = (half_t*)alloc(2 * 9 * 512 * 2);
  half_t* P_c2   = (half_t*)alloc(2 * 9 * 512 * 2);
  half_t* P_f3   = (half_t*)alloc(2 * 1 * 512 * 2);
  half_t* P_c3   = (half_t*)alloc(2 * 1 * 512 * 2);
  half_t* P_feat = (half_t*)alloc(2 * 3 * 512 * 2);
  half_t* P_zr   = (half_t*)alloc(4 * 27 * 512 * 2);
  half_t* P_q    = (half_t*)alloc(2 * 27 * 512 * 2);
  half_t* P_d1   = (half_t*)alloc(4 * 9 * 512 * 2);

  const int H1 = 192, W1 = 320, H = 96, W = 160;
  float*  t1     = (float*)alloc((size_t)32 * H1 * W1 * 4);
  half_t* t1n    = (half_t*)alloc(PAD1_HALVES * 2);      // halo-padded
  float*  t2     = (float*)alloc((size_t)32 * H * W * 4);
  half_t* t2n    = (half_t*)alloc((size_t)H * W * 32 * 2);
  half_t* fm1    = (half_t*)alloc((size_t)H * W * 32 * 2);
  half_t* fm2    = (half_t*)alloc((size_t)H * W * 32 * 2);
  float*  hidden = (float*)alloc((size_t)H * W * 32 * 4);
  half_t* h16    = (half_t*)alloc(PAD_HALVES * 2);
  half_t* feat16 = (half_t*)alloc(PAD_HALVES * 2);
  half_t* rh16   = (half_t*)alloc(PAD_HALVES * 2);
  float*  z32    = (float*)alloc((size_t)H * W * 32 * 4);
  half_t* hid16  = (half_t*)alloc(PAD_HALVES * 2);
  half_t* d1_16  = (half_t*)alloc((size_t)H * W * 64 * 2);

  auto cdiv = [](int a, int b) { return (a + b - 1) / b; };

  // ---- weight repacking ----
  auto pack = [&](const float* Wp, half_t* Pp, int M, int Cin, int KH, int KW,
                  int KchTot, int cOff) {
    int total = (M >> 4) * (KH * KW * (Cin >> 5)) * 512;
    pack_w_kernel<<<cdiv(total, 256), 256, 0, stream>>>(Wp, Pp, M, Cin, KH, KW, KchTot, cOff);
  };
  pack(fnet_w2, P_f2, 32, 32, 3, 3, 9, 0);
  pack(cnet_w2, P_c2, 32, 32, 3, 3, 9, 0);
  pack(fnet_w3, P_f3, 32, 32, 1, 1, 1, 0);
  pack(cnet_w3, P_c3, 32, 32, 1, 1, 1, 0);
  pack(bform_a,  P_feat, 32, 32, 1, 1, 3, 0);
  pack(bform_b,  P_feat, 32, 32, 1, 1, 3, 1);
  pack(bform_cd, P_feat, 32, 32, 1, 1, 3, 2);
  pack(gru_wz, P_zr, 32, 96, 3, 3, 27, 0);
  pack(gru_wr, P_zr + (size_t)2 * 27 * 512, 32, 96, 3, 3, 27, 0);
  pack(gru_wq, P_q, 32, 96, 3, 3, 27, 0);
  pack(disp_w1, P_d1, 64, 32, 3, 3, 9, 0);

  // ---- zero halo-padded buffers once (borders stay zero forever) ----
  int pad4 = (int)(PAD_HALVES * 2 / 16);
  int pad14 = (int)(PAD1_HALVES * 2 / 16);
  zero_kernel<<<cdiv(pad14, 256), 256, 0, stream>>>((uint4*)t1n, pad14);
  zero_kernel<<<cdiv(pad4, 256), 256, 0, stream>>>((uint4*)h16, pad4);
  zero_kernel<<<cdiv(pad4, 256), 256, 0, stream>>>((uint4*)feat16, pad4);
  zero_kernel<<<cdiv(pad4, 256), 256, 0, stream>>>((uint4*)rh16, pad4);
  zero_kernel<<<cdiv(pad4, 256), 256, 0, stream>>>((uint4*)hid16, pad4);

  // ---- encoders ----
  auto encoder = [&](const float* img, const float* w1, const float* b1,
                     half_t* Pc2, const float* b2, half_t* Pc3, const float* b3,
                     half_t* out16, float* out32, int mode) {
    conv7_s2_kernel<<<cdiv(32 * H1 * W1, 256), 256, 0, stream>>>(img, w1, b1, t1, 384, 640, H1, W1);
    inorm_relu_kernel<<<32, 256, 0, stream>>>(t1, t1n, H1 * W1, W1, 322, 1);
    conv2_s2_wmma_kernel<<<cdiv(H * (W / 16), 8), 256, 2 * 9 * 512 * 2, stream>>>(t1n, Pc2, b2, t2, H, W);
    inorm_relu_kernel<<<32, 256, 0, stream>>>(t2, t2n, H * W, W, W, 0);
    conv1x1_wmma_kernel<<<cdiv(H * W / 16, 8), 256, 0, stream>>>(t2n, Pc3, b3, out16, out32, H * W, mode);
  };
  encoder(image1, fnet_w1, fnet_b1, P_f2, fnet_b2, P_f3, fnet_b3, fm1, nullptr, 0);
  encoder(image2, fnet_w1, fnet_b1, P_f2, fnet_b2, P_f3, fnet_b3, fm2, nullptr, 0);
  encoder(image1, cnet_w1, cnet_b1, P_c2, cnet_b2, P_c3, cnet_b3, nullptr, hidden, 1);

  // ---- sequential GRU disparity sweep ----
  for (int d = 0; d < 48; ++d) {
    int Wd = W - d;
    int XTe = (Wd / 16) + 1;        // edge-zeroing tile count (feat, zr)
    int XTq = (Wd + 15) / 16;       // valid tile count (q)
    int blkE = cdiv(96 * XTe, 8);
    int blkQ = cdiv(96 * XTq, 8);
    feat_wmma_kernel<<<blkE, 256, 0, stream>>>(fm1, fm2, hidden, P_feat, cd_b, h16, feat16, d);
    zr_wmma_kernel<<<blkE, 256, 4 * 27 * 512 * 2, stream>>>(h16, feat16, P_zr, gru_bz, gru_br, z32, rh16, d);
    q_wmma_kernel<<<blkQ, 256, 2 * 27 * 512 * 2, stream>>>(rh16, h16, feat16, P_q, gru_bq, z32, hidden, d);
  }

  // ---- disparity head ----
  cvt_h16_kernel<<<cdiv(H * W * 32, 256), 256, 0, stream>>>(hidden, hid16);
  disp1_wmma_kernel<<<cdiv(96 * 10, 8), 256, 4 * 9 * 512 * 2, stream>>>(hid16, P_d1, disp_b1, d1_16);
  disp2_kernel<<<cdiv(H * W, 256), 256, 0, stream>>>(d1_16, disp_w2, disp_b2, (float*)d_out);
}
